// MultiHeadAttention_6210522710464
// MI455X (gfx1250) — matrix operand order repaired and checked by reading, not compiled, not run on hardware
//
#include <hip/hip_runtime.h>
#include <hip/hip_bf16.h>

#define Bb  4
#define Ss  2048
#define Hh  16
#define Dd  64
#define DMw 1024
#define EMBw 1024

typedef __bf16  v16bf __attribute__((ext_vector_type(16)));
typedef float   v8f   __attribute__((ext_vector_type(8)));
typedef unsigned int u32x4 __attribute__((ext_vector_type(4)));
typedef int          i32x8 __attribute__((ext_vector_type(8)));
typedef int          i32x4 __attribute__((ext_vector_type(4)));

union AV { v16bf v; unsigned int u[8]; };

// ---------------------------------------------------------------- TDM 2D tile load (bf16)
// Builds a Tensor DMA Descriptor (D#) per CDNA5 ISA §8.3/8.4 and issues
// tensor_load_to_lds (6-arg toolchain form). tile_x elems contiguous in
// memory, tile_y rows with row_stride elems between rows. Tracked by TENSORcnt.
__device__ __forceinline__ void tdm_load_tile_bf16(unsigned lds_off, const void* gptr,
                                                   unsigned tile_x, unsigned tile_y,
                                                   unsigned row_stride) {
    unsigned long long ga = (unsigned long long)gptr;
    u32x4 g0;
    g0[0] = 1u;                                   // count=1, no gather, user mode
    g0[1] = lds_off;                              // LDS byte address
    g0[2] = (unsigned)(ga & 0xffffffffull);       // global_addr[31:0]
    g0[3] = (unsigned)((ga >> 32) & 0x01ffffffull) | (2u << 30);  // addr[56:32] | type=2
    i32x8 g1;
    g1[0] = (int)(1u << 16);                      // data_size=1 (2 bytes/elem)
    g1[1] = (int)((tile_x & 0xffffu) << 16);      // tensor_dim0[15:0] (== tile_x)
    g1[2] = (int)(((tile_x >> 16) & 0xffffu) | ((tile_y & 0xffffu) << 16)); // dim0 hi | dim1 lo
    g1[3] = (int)(((tile_y >> 16) & 0xffffu) | ((tile_x & 0xffffu) << 16)); // dim1 hi | tile_dim0
    g1[4] = (int)(tile_y & 0xffffu);              // tile_dim1; tile_dim2=0
    g1[5] = (int)row_stride;                      // tensor_dim0_stride[31:0]
    g1[6] = 0;                                    // stride0 hi | stride1 lo
    g1[7] = 0;
    i32x4 z4 = {0, 0, 0, 0};
    i32x8 z8 = {0, 0, 0, 0, 0, 0, 0, 0};
    __builtin_amdgcn_tensor_load_to_lds(g0, g1, z4, z4, z8, 0);
}

__device__ __forceinline__ unsigned lds_offset_of(const void* p) {
    // LDS aperture: generic addr[31:0] == wave-relative LDS byte offset
    return (unsigned)(unsigned long long)p;
}

// ---------------------------------------------------------------- convert f32 -> bf16
__global__ __launch_bounds__(256)
void cvt_f32_bf16(const float* __restrict__ src, __bf16* __restrict__ dst, int n) {
    int i = (blockIdx.x * 256 + threadIdx.x) * 4;
    if (i + 3 < n) {
        float4 f = *(const float4*)(src + i);
        dst[i + 0] = (__bf16)f.x; dst[i + 1] = (__bf16)f.y;
        dst[i + 2] = (__bf16)f.z; dst[i + 3] = (__bf16)f.w;
    } else {
        for (; i < n; ++i) dst[i] = (__bf16)src[i];
    }
}

// -------------------------------------------- transpose + convert: W[k][n] -> Wt[n][k] bf16
__global__ __launch_bounds__(256)
void transpose_f32_bf16(const float* __restrict__ src, __bf16* __restrict__ dst, int K, int N) {
    __shared__ float tile[32][33];
    int k0 = blockIdx.y * 32, n0 = blockIdx.x * 32;
    int tx = threadIdx.x & 31, ty = threadIdx.x >> 5;   // ty 0..7
    #pragma unroll
    for (int r = ty; r < 32; r += 8)
        tile[r][tx] = src[(size_t)(k0 + r) * N + n0 + tx];
    __syncthreads();
    #pragma unroll
    for (int r = ty; r < 32; r += 8)
        dst[(size_t)(n0 + r) * K + k0 + tx] = (__bf16)tile[tx][r];
}

// ---------------------------------------------------------------- bf16 GEMM, A[M,K] * Bt[N,K]^T
// Tiles staged by the Tensor Data Mover, double buffered; 8 waves x (16x64) WMMA tiles.
// MODE 0: out = bf16 head-split [B,H,S,D] with bias (QKV projections)
// MODE 1: out = f32 row-major [M,N] (final projection)
template <int MODE>
__global__ __launch_bounds__(256)
void gemm64x128(const __bf16* __restrict__ A, const __bf16* __restrict__ Bt,
                const float* __restrict__ bias, __bf16* __restrict__ outb,
                float* __restrict__ outf, int M, int N, int K) {
    __shared__ __align__(16) __bf16 lA[2][64][64];
    __shared__ __align__(16) __bf16 lB[2][128][64];
    const int tid  = threadIdx.x;
    const int l    = tid & 31, w = tid >> 5;
    const int wy   = w >> 1,  wx = w & 1;
    const int ln   = l & 15,  hsel = l >> 4;
    const int m0   = blockIdx.y * 64, n0 = blockIdx.x * 128;
    const int am   = wy * 16 + ln;

    v8f acc[4] = {};

    // prologue: DMA first tiles
    if (w == 0) {
        tdm_load_tile_bf16(lds_offset_of(&lA[0][0][0]), A + (size_t)m0 * K, 64, 64, K);
        tdm_load_tile_bf16(lds_offset_of(&lB[0][0][0]), Bt + (size_t)n0 * K, 64, 128, K);
        __builtin_amdgcn_s_wait_tensorcnt(0);
    }
    __syncthreads();

    int cur = 0;
    for (int k0 = 0; k0 < K; k0 += 64) {
        const int nxt = cur ^ 1;
        const bool more = (k0 + 64) < K;
        if (more && w == 0) {   // overlap next-tile DMA with this tile's WMMAs
            tdm_load_tile_bf16(lds_offset_of(&lA[nxt][0][0]),
                               A + (size_t)m0 * K + (k0 + 64), 64, 64, K);
            tdm_load_tile_bf16(lds_offset_of(&lB[nxt][0][0]),
                               Bt + (size_t)n0 * K + (k0 + 64), 64, 128, K);
        }

        #pragma unroll
        for (int ks = 0; ks < 64; ks += 32) {
            AV av;
            #pragma unroll
            for (int i = 0; i < 8; ++i) {
                int kk = ks + ((i >> 2) * 16) + hsel * 8 + ((i & 3) * 2);
                av.u[i] = *(const unsigned int*)&lA[cur][am][kk];
            }
            #pragma unroll
            for (int t = 0; t < 4; ++t) {
                AV bv;
                int nn = wx * 64 + t * 16 + ln;
                #pragma unroll
                for (int i = 0; i < 8; ++i) {
                    int kk = ks + ((i >> 2) * 16) + hsel * 8 + ((i & 3) * 2);
                    bv.u[i] = *(const unsigned int*)&lB[cur][nn][kk];
                }
                acc[t] = __builtin_amdgcn_wmma_f32_16x16x32_bf16(
                    false, av.v, false, bv.v, (short)0, acc[t], false, false);
            }
        }
        __syncthreads();                       // all waves done reading `cur`
        if (more && w == 0) __builtin_amdgcn_s_wait_tensorcnt(0);
        __syncthreads();                       // `nxt` tiles visible to all waves
        cur = nxt;
    }

    // epilogue
    #pragma unroll
    for (int t = 0; t < 4; ++t) {
        int ng = n0 + wx * 64 + t * 16 + ln;
        float bv = (MODE == 0) ? bias[ng] : 0.0f;
        #pragma unroll
        for (int v = 0; v < 8; ++v) {
            int mg = m0 + wy * 16 + hsel * 8 + v;
            float val = acc[t][v] + bv;
            if (MODE == 0) {
                int b = mg >> 11;          // / Ss
                int s = mg & (Ss - 1);
                int h = ng >> 6;           // / Dd
                int d = ng & (Dd - 1);
                outb[(((size_t)(b * Hh + h)) * Ss + s) * Dd + d] = (__bf16)val;
            } else {
                outf[(size_t)mg * N + ng] = val;
            }
        }
    }
}

// ---------------------------------------------------------------- flash attention
// grid: (S/64, H, B), block: 128 (4 waves x 16 q-rows)
__global__ __launch_bounds__(128)
void attn_flash(const __bf16* __restrict__ Q, const __bf16* __restrict__ K,
                const __bf16* __restrict__ V, const float* __restrict__ mask,
                __bf16* __restrict__ C) {
    __shared__ __align__(16) __bf16 lK[64][64];    // [key][d]  (TDM-staged)
    __shared__ __align__(16) __bf16 lVt[64][64];   // [d][kv]   (manual transpose)
    __shared__ __align__(16) __bf16 lP[4][16][64]; // per-wave P tile

    const int tid  = threadIdx.x;
    const int l    = tid & 31, w = tid >> 5;
    const int ln   = l & 15,  hsel = l >> 4;
    const int b    = blockIdx.z, h = blockIdx.y;
    const int q0   = blockIdx.x * 64 + w * 16;
    const size_t headoff = ((size_t)(b * Hh + h)) * Ss * Dd;
    const __bf16* Qh = Q + headoff;
    const __bf16* Kh = K + headoff;
    const __bf16* Vh = V + headoff;

    // Q A-operands for both 32-wide halves of D=64 (persistent in registers)
    AV aQ[2];
    {
        int m = q0 + ln;
        #pragma unroll
        for (int hd = 0; hd < 2; ++hd)
            #pragma unroll
            for (int i = 0; i < 8; ++i) {
                int kk = hd * 32 + ((i >> 2) * 16) + hsel * 8 + ((i & 3) * 2);
                aQ[hd].u[i] = *(const unsigned int*)&Qh[(size_t)m * Dd + kk];
            }
    }

    float mrun[8], lrun[8];
    #pragma unroll
    for (int v = 0; v < 8; ++v) { mrun[v] = -3.0e38f; lrun[v] = 0.0f; }
    v8f acc[4] = {};
    const float scl = 0.125f;   // 1/sqrt(64)

    for (int kv0 = 0; kv0 < Ss; kv0 += 64) {
        __syncthreads();       // previous iteration's reads of lK/lVt complete
        // K tile: Tensor Data Mover (overlaps with V staging below)
        if (w == 0)
            tdm_load_tile_bf16(lds_offset_of(&lK[0][0]),
                               Kh + (size_t)kv0 * Dd, 64, 64, Dd);
        // V tile: manual transposed staging [d][kv]
        #pragma unroll
        for (int j = 0; j < 4; ++j) {
            int c = tid * 4 + j;
            int r = c >> 3, col = (c & 7) * 8;
            uint4 vv = *(const uint4*)&Vh[(size_t)(kv0 + r) * Dd + col];
            const __bf16* ve = (const __bf16*)&vv;
            #pragma unroll
            for (int e = 0; e < 8; ++e) lVt[col + e][r] = ve[e];
        }
        if (w == 0) __builtin_amdgcn_s_wait_tensorcnt(0);
        __syncthreads();

        // scores S = Q K^T : 4 key subtiles x 2 d-halves
        v8f sc[4];
        #pragma unroll
        for (int t = 0; t < 4; ++t) {
            v8f z = {};
            sc[t] = z;
            int key = t * 16 + ln;
            #pragma unroll
            for (int hd = 0; hd < 2; ++hd) {
                AV bK;
                #pragma unroll
                for (int i = 0; i < 8; ++i) {
                    int dd = hd * 32 + ((i >> 2) * 16) + hsel * 8 + ((i & 3) * 2);
                    bK.u[i] = *(const unsigned int*)&lK[key][dd];
                }
                sc[t] = __builtin_amdgcn_wmma_f32_16x16x32_bf16(
                    false, aQ[hd].v, false, bK.v, (short)0, sc[t], false, false);
            }
        }

        // scale + mask (reference: softmax(att*scale - mask))
        #pragma unroll
        for (int t = 0; t < 4; ++t) {
            int col = kv0 + t * 16 + ln;
            #pragma unroll
            for (int v = 0; v < 8; ++v) {
                int row = q0 + hsel * 8 + v;
                float mval = mask[(size_t)b * Ss * Ss + (size_t)row * Ss + col];
                sc[t][v] = sc[t][v] * scl - mval;
            }
        }

        // online softmax, write P (bf16) to per-wave LDS
        #pragma unroll
        for (int v = 0; v < 8; ++v) {
            float rm = fmaxf(fmaxf(sc[0][v], sc[1][v]), fmaxf(sc[2][v], sc[3][v]));
            #pragma unroll
            for (int off = 1; off < 16; off <<= 1)
                rm = fmaxf(rm, __shfl_xor(rm, off, 16));
            float mn    = fmaxf(mrun[v], rm);
            float alpha = __expf(mrun[v] - mn);
            float rs = 0.0f;
            #pragma unroll
            for (int t = 0; t < 4; ++t) {
                float p = __expf(sc[t][v] - mn);
                rs += p;
                lP[w][hsel * 8 + v][t * 16 + ln] = (__bf16)p;
            }
            #pragma unroll
            for (int off = 1; off < 16; off <<= 1)
                rs += __shfl_xor(rs, off, 16);
            lrun[v] = lrun[v] * alpha + rs;
            mrun[v] = mn;
            #pragma unroll
            for (int t = 0; t < 4; ++t) acc[t][v] *= alpha;
        }

        // ctx += P * V : 2 kv-halves x 4 d subtiles
        #pragma unroll
        for (int c = 0; c < 2; ++c) {
            AV aP;
            #pragma unroll
            for (int i = 0; i < 8; ++i) {
                int kk = c * 32 + ((i >> 2) * 16) + hsel * 8 + ((i & 3) * 2);
                aP.u[i] = *(const unsigned int*)&lP[w][ln][kk];
            }
            #pragma unroll
            for (int t = 0; t < 4; ++t) {
                AV bV;
                int dd = t * 16 + ln;
                #pragma unroll
                for (int i = 0; i < 8; ++i) {
                    int kk = c * 32 + ((i >> 2) * 16) + hsel * 8 + ((i & 3) * 2);
                    bV.u[i] = *(const unsigned int*)&lVt[dd][kk];
                }
                acc[t] = __builtin_amdgcn_wmma_f32_16x16x32_bf16(
                    false, aP.v, false, bV.v, (short)0, acc[t], false, false);
            }
        }
    }

    // finalize: ctx/l, write [B,S, H*D] bf16 for the output projection
    #pragma unroll
    for (int t = 0; t < 4; ++t) {
        int dd = t * 16 + ln;
        #pragma unroll
        for (int v = 0; v < 8; ++v) {
            int srow = q0 + hsel * 8 + v;
            float val = acc[t][v] / lrun[v];
            C[((size_t)b * Ss + srow) * DMw + h * Dd + dd] = (__bf16)val;
        }
    }
}

// ---------------------------------------------------------------- host launcher
extern "C" void kernel_launch(void* const* d_in, const int* in_sizes, int n_in,
                              void* d_out, int out_size, void* d_ws, size_t ws_size,
                              hipStream_t stream) {
    const float* X    = (const float*)d_in[0];
    const float* mask = (const float*)d_in[1];
    const float* Wq   = (const float*)d_in[2];
    const float* bq   = (const float*)d_in[3];
    const float* Wk   = (const float*)d_in[4];
    const float* bk   = (const float*)d_in[5];
    const float* Wv   = (const float*)d_in[6];
    const float* bv   = (const float*)d_in[7];
    const float* Wo   = (const float*)d_in[8];
    float* out        = (float*)d_out;

    const int M  = Bb * Ss;          // 8192
    const int Kd = DMw;              // 1024
    const int Nq = Hh * Dd;          // 1024

    char* ws = (char*)d_ws;
    size_t off = 0;
    auto carve = [&](size_t bytes) { char* p = ws + off; off += (bytes + 255) & ~(size_t)255; return p; };

    __bf16* X16 = (__bf16*)carve((size_t)M * Kd * 2);
    __bf16* WqT = (__bf16*)carve((size_t)Kd * Nq * 2);
    __bf16* WkT = (__bf16*)carve((size_t)Kd * Nq * 2);
    __bf16* WvT = (__bf16*)carve((size_t)Kd * Nq * 2);
    __bf16* WoT = (__bf16*)carve((size_t)Kd * EMBw * 2);
    __bf16* Q16 = (__bf16*)carve((size_t)M * Nq * 2);
    __bf16* K16 = (__bf16*)carve((size_t)M * Nq * 2);
    __bf16* V16 = (__bf16*)carve((size_t)M * Nq * 2);
    __bf16* C16 = (__bf16*)carve((size_t)M * Nq * 2);

    // 1) convert X to bf16
    {
        int n = M * Kd;
        cvt_f32_bf16<<<dim3((n / 4 + 255) / 256), dim3(256), 0, stream>>>(X, X16, n);
    }
    // 2) transpose+convert weights
    {
        dim3 g(Nq / 32, Kd / 32), bdim(256);
        transpose_f32_bf16<<<g, bdim, 0, stream>>>(Wq, WqT, Kd, Nq);
        transpose_f32_bf16<<<g, bdim, 0, stream>>>(Wk, WkT, Kd, Nq);
        transpose_f32_bf16<<<g, bdim, 0, stream>>>(Wv, WvT, Kd, Nq);
        transpose_f32_bf16<<<g, bdim, 0, stream>>>(Wo, WoT, Kd, EMBw);
    }
    // 3) QKV projections -> head-split bf16
    {
        dim3 g(Nq / 128, M / 64), bdim(256);
        gemm64x128<0><<<g, bdim, 0, stream>>>(X16, WqT, bq, Q16, nullptr, M, Nq, Kd);
        gemm64x128<0><<<g, bdim, 0, stream>>>(X16, WkT, bk, K16, nullptr, M, Nq, Kd);
        gemm64x128<0><<<g, bdim, 0, stream>>>(X16, WvT, bv, V16, nullptr, M, Nq, Kd);
    }
    // 4) flash attention -> ctx bf16 [B,S,H*D]
    {
        dim3 g(Ss / 64, Hh, Bb), bdim(128);
        attn_flash<<<g, bdim, 0, stream>>>(Q16, K16, V16, mask, C16);
    }
    // 5) output projection -> f32 d_out
    {
        dim3 g(EMBw / 128, M / 64), bdim(256);
        gemm64x128<1><<<g, bdim, 0, stream>>>(C16, WoT, nullptr, nullptr, out, M, EMBw, Kd);
    }
}
